// CustomEmbedding_52140902973622
// MI455X (gfx1250) — compile-verified, matrix operand order checked by reading, not run on hardware
//
#include <hip/hip_runtime.h>
#include <hip/hip_bf16.h>

typedef __attribute__((ext_vector_type(16))) _Float16 v16h;
typedef __attribute__((ext_vector_type(8)))  float    v8f;

#define SQ    2048   // sequence length S
#define BATCH 8      // batch B
#define NTOK  64     // vocabulary
#define DM    648    // d_model
#define DMAXD 4      // max prefix depth

// -------------------------------------------------------------------------
// Kernel 1: elementwise channels.
//   cc in [0,320)  -> one-hot blocks (offset m = cc>>6, token c = cc&63)
//   cc == 320      -> channel 644 = q counter
//   cc in 321..323 -> channels 645..647 = 0 (pad)
// -------------------------------------------------------------------------
__global__ void fill_kernel(const int* __restrict__ src, float* __restrict__ out) {
    const int CC = 324;
    long long idx = (long long)blockIdx.x * blockDim.x + threadIdx.x;
    const long long total = (long long)SQ * BATCH * CC;
    if (idx >= total) return;
    int cc = (int)(idx % CC);
    int b  = (int)((idx / CC) % BATCH);
    int t  = (int)(idx / ((long long)CC * BATCH));
    int ch;
    float val;
    if (cc < 320) {
        ch = cc;
        int m = cc >> 6, c = cc & 63;
        val = (t >= m && src[(t - m) * BATCH + b] == c) ? 1.0f : 0.0f;
    } else if (cc == 320) {
        ch = 644; val = (float)t;
    } else {
        ch = 644 + (cc - 320); val = 0.0f;   // 645..647 pad
    }
    out[(size_t)t * (BATCH * DM) + (size_t)b * DM + ch] = val;
}

// -------------------------------------------------------------------------
// Kernel 2: packed contexts. ctx[t*B+b] byte k = src[t-k] (0xFF if t-k < 0).
// Depth-i match(q, j): ((ctx[q] ^ ctx[j-1]) & ((1<<8i)-1)) == 0.
// Byte 0 doubles as the token for the one-hot B operand.
// -------------------------------------------------------------------------
__global__ void ctx_kernel(const int* __restrict__ src, unsigned int* __restrict__ ctx) {
    int idx = blockIdx.x * blockDim.x + threadIdx.x;
    if (idx >= SQ * BATCH) return;
    int b = idx % BATCH, t = idx / BATCH;
    unsigned int c = 0u;
#pragma unroll
    for (int k = 0; k < DMAXD; ++k) {
        unsigned int byte = (t - k >= 0) ? (unsigned int)(src[(t - k) * BATCH + b] & 0xFF)
                                         : 0xFFu;
        c |= byte << (8 * k);
    }
    ctx[idx] = c;
}

// -------------------------------------------------------------------------
// Kernel 3: exclusive causal one-hot mean (channels 576..639).
// One block per batch column, one thread per token id; sequential scan.
// -------------------------------------------------------------------------
__global__ __launch_bounds__(64) void scan_kernel(const int* __restrict__ src,
                                                  float* __restrict__ out) {
    int b = blockIdx.x;
    int c = threadIdx.x;             // token id 0..63
    float cnt = 0.0f;
    for (int t = 0; t < SQ; ++t) {
        float denom = (t >= 1) ? (float)t : 1.0f;
        out[(size_t)t * (BATCH * DM) + (size_t)b * DM + 576 + c] = cnt / denom;
        if (src[t * BATCH + b] == c) cnt += 1.0f;
    }
}

// -------------------------------------------------------------------------
// Kernel 4: WMMA prefix-match histogram.
// One wave (32 threads) per (16-row q-tile, batch). For each 32-wide j-tile:
//   A_i[16x32] f16 : depth-i match & (j<q) built from packed-context XORs
//   B_n[32x16] f16 : one-hot columns n*16..n*16+15 from token byte
//   acc[i][n] += A_i * B_n  via v_wmma_f32_16x16x32_f16   (16 WMMA / K-step)
// Then counts_i[q] = row-sum via shfl-xor reduction, averages = acc/denom.
//
// Fragment layouts (wave32, per ISA 7.12.2):
//   A: lane L -> row M = L&15; K pairs per VGPR v: base(v) = (v<4?2v:16+2(v-4)) + 8*(L>>4)
//   B: lane L -> col N = L&15; same K striping
//   C: lane L -> col N = L&15; VGPR v -> row M = v + 8*(L>>4)
// -------------------------------------------------------------------------
__global__ __launch_bounds__(32) void prefix_wmma_kernel(const unsigned int* __restrict__ ctx,
                                                         float* __restrict__ out) {
    __shared__ unsigned int jctx[33];      // [0] = ctx[jbase-1], [1+k] = ctx[jbase+k]
    const int lane  = threadIdx.x;         // 0..31
    const int qbase = blockIdx.x * 16;     // q-tile
    const int b     = blockIdx.y;          // batch
    const int g     = lane >> 4;           // which K-half this lane holds
    const int nn    = lane & 15;           // N (B/C) or M (A) index

    const int q_of_lane = qbase + nn;                         // A-row position
    const unsigned int ctxq = ctx[q_of_lane * BATCH + b];     // q-side context

    v8f acc[DMAXD][4];
#pragma unroll
    for (int i = 0; i < DMAXD; ++i)
#pragma unroll
        for (int n = 0; n < 4; ++n)
            acc[i][n] = (v8f){0.f, 0.f, 0.f, 0.f, 0.f, 0.f, 0.f, 0.f};

    const unsigned int dmask[DMAXD] = {0xFFu, 0xFFFFu, 0xFFFFFFu, 0xFFFFFFFFu};

    for (int jbase = 0; jbase < qbase + 16; jbase += 32) {
        // Stage this j-tile's contexts in LDS (ds_store/ds_load traffic).
        jctx[1 + lane] = ctx[(jbase + lane) * BATCH + b];
        if (lane == 0)
            jctx[0] = (jbase > 0) ? ctx[(jbase - 1) * BATCH + b] : 0xFFFFFFFFu;
        // Prefetch next tile's contexts toward L2 while we compute.
        if (jbase + 32 < qbase + 16)
            __builtin_prefetch(&ctx[(jbase + 32 + lane) * BATCH + b], 0, 1);
        __syncthreads();

        // ---- A fragments: one per depth, shared XOR per K element ----
        v16h afrag[DMAXD];
#pragma unroll
        for (int v = 0; v < 8; ++v) {
            const int k0 = ((v < 4) ? 2 * v : 16 + 2 * (v - 4)) + 8 * g;
            const int k1 = k0 + 1;
            const unsigned int x0 = ctxq ^ jctx[k0];       // ctx[j0-1]
            const unsigned int x1 = ctxq ^ jctx[k1];       // ctx[j1-1]
            const bool in0 = (jbase + k0) < q_of_lane;     // causal mask j < q
            const bool in1 = (jbase + k1) < q_of_lane;
#pragma unroll
            for (int i = 0; i < DMAXD; ++i) {
                afrag[i][2 * v]     = (in0 && ((x0 & dmask[i]) == 0u)) ? (_Float16)1.0f
                                                                       : (_Float16)0.0f;
                afrag[i][2 * v + 1] = (in1 && ((x1 & dmask[i]) == 0u)) ? (_Float16)1.0f
                                                                       : (_Float16)0.0f;
            }
        }

        // ---- per N-tile: one-hot B fragment, then 4 depth WMMAs ----
#pragma unroll
        for (int n = 0; n < 4; ++n) {
            const unsigned int col = (unsigned int)(n * 16 + nn);
            v16h bfrag;
#pragma unroll
            for (int v = 0; v < 8; ++v) {
                const int k0 = ((v < 4) ? 2 * v : 16 + 2 * (v - 4)) + 8 * g;
                const unsigned int t0 = jctx[1 + k0] & 0xFFu;      // token at j0
                const unsigned int t1 = jctx[2 + k0] & 0xFFu;      // token at j1
                bfrag[2 * v]     = (t0 == col) ? (_Float16)1.0f : (_Float16)0.0f;
                bfrag[2 * v + 1] = (t1 == col) ? (_Float16)1.0f : (_Float16)0.0f;
            }
#pragma unroll
            for (int i = 0; i < DMAXD; ++i) {
                acc[i][n] = __builtin_amdgcn_wmma_f32_16x16x32_f16(
                    false, afrag[i], false, bfrag, (short)0, acc[i][n], false, false);
            }
        }
        __syncthreads();
    }

    // ---- counts (row-sums) via shfl-xor, then averaged stores ----
#pragma unroll
    for (int i = 0; i < DMAXD; ++i) {
#pragma unroll
        for (int v = 0; v < 8; ++v) {
            float p = acc[i][0][v] + acc[i][1][v] + acc[i][2][v] + acc[i][3][v];
            p += __shfl_xor(p, 1, 32);
            p += __shfl_xor(p, 2, 32);
            p += __shfl_xor(p, 4, 32);
            p += __shfl_xor(p, 8, 32);        // all 16 lanes of this half hold count
            const float denom = (p > 1.0f) ? p : 1.0f;
            const int qrow = qbase + v + 8 * g;
            float* rowp = out + (size_t)qrow * (BATCH * DM) + (size_t)b * DM;
#pragma unroll
            for (int n = 0; n < 4; ++n)
                rowp[320 + i * 64 + n * 16 + nn] = acc[i][n][v] / denom;
            if (nn == 0)
                rowp[640 + i] = p;            // depth-(i+1) match count
        }
    }
}

// -------------------------------------------------------------------------
extern "C" void kernel_launch(void* const* d_in, const int* in_sizes, int n_in,
                              void* d_out, int out_size, void* d_ws, size_t ws_size,
                              hipStream_t stream) {
    const int* src = (const int*)d_in[0];          // [S, B] int32
    float* out = (float*)d_out;                    // [S, B, 648] float32
    unsigned int* ctx = (unsigned int*)d_ws;       // S*B u32 packed contexts (64 KB)

    // 1) elementwise channels (one-hots, counter, pad)
    {
        long long total = (long long)SQ * BATCH * 324;
        int blocks = (int)((total + 255) / 256);
        fill_kernel<<<blocks, 256, 0, stream>>>(src, out);
    }
    // 2) packed contexts
    {
        int total = SQ * BATCH;
        ctx_kernel<<<(total + 255) / 256, 256, 0, stream>>>(src, ctx);
    }
    // 3) exclusive causal mean
    scan_kernel<<<BATCH, 64, 0, stream>>>(src, out);
    // 4) WMMA prefix-match averages + counts
    {
        dim3 grid(SQ / 16, BATCH);
        prefix_wmma_kernel<<<grid, 32, 0, stream>>>(ctx, out);
    }
}
